// MIOECrossAttentionBlock_13726715478156
// MI455X (gfx1250) — matrix operand order repaired and checked by reading, not compiled, not run on hardware
//
#include <hip/hip_runtime.h>
#include <hip/hip_bf16.h>

typedef _Float16 h16;
typedef _Float16 half8 __attribute__((ext_vector_type(8)));
typedef _Float16 v16h  __attribute__((ext_vector_type(16)));
typedef float    v8f   __attribute__((ext_vector_type(8)));

static constexpr int Bn  = 8;
static constexpr int T1n = 4096;
static constexpr int T2n = 2048;
static constexpr int Cn  = 256;
static constexpr int Hn  = 8;
static constexpr int NIn = 512;
static constexpr int M1n = Bn * T1n;   // 32768
static constexpr int M2n = Bn * T2n;   // 16384

// ---------------------------------------------------------------- helpers
__device__ __forceinline__ float gelu_f(float v) {
    return 0.5f * v * (1.0f + erff(v * 0.7071067811865475f));
}
__device__ __forceinline__ float wave_sum(float v) {
    #pragma unroll
    for (int o = 16; o; o >>= 1) v += __shfl_xor(v, o, 32);
    return v;
}
__device__ __forceinline__ float wave_max(float v) {
    #pragma unroll
    for (int o = 16; o; o >>= 1) v = fmaxf(v, __shfl_xor(v, o, 32));
    return v;
}
__device__ __forceinline__ v8f wmma16(v16h a, v16h b, v8f c) {
    return __builtin_amdgcn_wmma_f32_16x16x32_f16(false, a, false, b, (short)0, c, false, false);
}

// ---------------------------------------------------------------- weight transpose: f32[K,N] -> f16[N,K]
__global__ void wtrans_kernel(const float* __restrict__ src, h16* __restrict__ dst, int K, int N) {
    int idx = blockIdx.x * 256 + threadIdx.x;
    if (idx >= K * N) return;
    int k = idx / N, n = idx % N;
    dst[(size_t)n * K + k] = (h16)src[idx];
}

// ---------------------------------------------------------------- generic WMMA GEMM
// out = epi(A[M,K]f16 @ WT[N,K]f16 + bias[N])
// block: 256 thr = 8 waves; block tile 128 rows x 128 cols; wave w: rows w*16, 8 n-tiles.
// EPI: 0 f32 store, 1 f16 store, 2 gelu->f16, 3 f32 +=, 4 f32 += gate[row*4+e]*v
template <int EPI>
__global__ __launch_bounds__(256) void gemm_wmma(
    const h16* __restrict__ A, const h16* __restrict__ WT, const float* __restrict__ bias,
    void* __restrict__ out, int M, int K, int N,
    const float* __restrict__ gate, int expert)
{
    int lane = threadIdx.x & 31;
    int wave = threadIdx.x >> 5;
    int m0 = blockIdx.x * 128 + wave * 16;
    int n0 = blockIdx.y * 128;
    int hsel = lane >> 4, l15 = lane & 15;
    const h16* arow = A + (size_t)(m0 + l15) * K;
    v8f acc[8];
    #pragma unroll
    for (int nt = 0; nt < 8; ++nt)
        #pragma unroll
        for (int r = 0; r < 8; ++r) acc[nt][r] = 0.0f;

    for (int k0 = 0; k0 < K; k0 += 32) {
        if (k0 + 32 < K)
            __builtin_prefetch((const void*)(arow + k0 + 32 + hsel * 8), 0, 1);
        half8 alo = *(const half8*)(arow + k0 + hsel * 16);
        half8 ahi = *(const half8*)(arow + k0 + hsel * 16 + 8);
        v16h af;
        #pragma unroll
        for (int j = 0; j < 8; ++j) { af[j] = alo[j]; af[j + 8] = ahi[j]; }
        #pragma unroll
        for (int nt = 0; nt < 8; ++nt) {
            const h16* bp = WT + (size_t)(n0 + nt * 16 + l15) * K + k0 + hsel * 16;
            v16h bf = *(const v16h*)bp;
            acc[nt] = wmma16(af, bf, acc[nt]);
        }
    }
    #pragma unroll
    for (int nt = 0; nt < 8; ++nt) {
        int ncol = n0 + nt * 16 + l15;
        float bv = bias[ncol];
        #pragma unroll
        for (int r = 0; r < 8; ++r) {
            int row = m0 + r + 8 * hsel;
            float v = acc[nt][r] + bv;
            size_t idx = (size_t)row * N + ncol;
            if (EPI == 0)      ((float*)out)[idx] = v;
            else if (EPI == 1) ((h16*)out)[idx] = (h16)v;
            else if (EPI == 2) ((h16*)out)[idx] = (h16)gelu_f(v);
            else if (EPI == 3) ((float*)out)[idx] += v;
            else if (EPI == 4) ((float*)out)[idx] += gate[(size_t)row * 4 + expert] * v;
        }
    }
}

// ---------------------------------------------------------------- LayerNorm (C=256), one wave per row
__global__ __launch_bounds__(256) void ln_f16_kernel(
    const float* __restrict__ src, const float* __restrict__ g, const float* __restrict__ b,
    h16* __restrict__ dst)
{
    int row = blockIdx.x * 8 + (threadIdx.x >> 5);
    int lane = threadIdx.x & 31;
    const float* sr = src + (size_t)row * Cn;
    float x[8]; float s = 0.0f;
    #pragma unroll
    for (int j = 0; j < 8; ++j) { x[j] = sr[lane + j * 32]; s += x[j]; }
    float mean = wave_sum(s) * (1.0f / Cn);
    float vs = 0.0f;
    #pragma unroll
    for (int j = 0; j < 8; ++j) { float d = x[j] - mean; vs += d * d; }
    float rstd = rsqrtf(wave_sum(vs) * (1.0f / Cn) + 1e-5f);
    #pragma unroll
    for (int j = 0; j < 8; ++j) {
        int c = lane + j * 32;
        dst[(size_t)row * Cn + c] = (h16)((x[j] - mean) * rstd * g[c] + b[c]);
    }
}

// dst = resid + LN(src)*g+b   (dst may alias resid)
__global__ __launch_bounds__(256) void ln_add_kernel(
    const float* __restrict__ src, const float* __restrict__ resid,
    const float* __restrict__ g, const float* __restrict__ b, float* __restrict__ dst)
{
    int row = blockIdx.x * 8 + (threadIdx.x >> 5);
    int lane = threadIdx.x & 31;
    const float* sr = src + (size_t)row * Cn;
    float x[8]; float s = 0.0f;
    #pragma unroll
    for (int j = 0; j < 8; ++j) { x[j] = sr[lane + j * 32]; s += x[j]; }
    float mean = wave_sum(s) * (1.0f / Cn);
    float vs = 0.0f;
    #pragma unroll
    for (int j = 0; j < 8; ++j) { float d = x[j] - mean; vs += d * d; }
    float rstd = rsqrtf(wave_sum(vs) * (1.0f / Cn) + 1e-5f);
    #pragma unroll
    for (int j = 0; j < 8; ++j) {
        int c = lane + j * 32;
        size_t idx = (size_t)row * Cn + c;
        dst[idx] = resid[idx] + (x[j] - mean) * rstd * g[c] + b[c];
    }
}

// ---------------------------------------------------------------- softmax over 32-channel head groups
__global__ __launch_bounds__(256) void softmax32_kernel(
    const float* __restrict__ src, h16* __restrict__ dst)
{
    int row = blockIdx.x;
    int h = threadIdx.x >> 5;
    int lane = threadIdx.x & 31;
    size_t idx = (size_t)row * Cn + h * 32 + lane;
    float v = src[idx];
    float m = wave_max(v);
    float e = __expf(v - m);
    float s = wave_sum(e);
    dst[idx] = (h16)(e / s);
}

// ---------------------------------------------------------------- ksum[d] = sum_t k[b,t,h*32+d]
__global__ __launch_bounds__(256) void ksum_kernel(
    const h16* __restrict__ k16, float* __restrict__ ksum, int T)
{
    int bh = blockIdx.x; int b = bh >> 3, h = bh & 7;
    int d = threadIdx.x & 31;
    int part = threadIdx.x >> 5;
    float s = 0.0f;
    for (int t = part; t < T; t += 8)
        s += (float)k16[(size_t)(b * T + t) * Cn + h * 32 + d];
    __shared__ float red[8][32];
    red[part][d] = s;
    __syncthreads();
    if (part == 0) {
        float tot = 0.0f;
        #pragma unroll
        for (int j = 0; j < 8; ++j) tot += red[j][d];
        ksum[bh * 32 + d] = tot;
    }
}

// ---------------------------------------------------------------- ctx = k^T v  (32x32 per b,h), WMMA over T
__global__ __launch_bounds__(128) void ctx_kernel(
    const h16* __restrict__ k16, const h16* __restrict__ v16,
    h16* __restrict__ ctxT, int T)
{
    int bh = blockIdx.x; int b = bh >> 3, h = bh & 7;
    __shared__ h16 kT[32][40];
    __shared__ h16 vT[32][40];
    int lane = threadIdx.x & 31;
    int wave = threadIdx.x >> 5;
    int m0 = (wave & 1) * 16;   // d rows
    int n0 = (wave >> 1) * 16;  // e cols
    int hsel = lane >> 4, l15 = lane & 15;
    int tr = threadIdx.x & 31;
    int seg = threadIdx.x >> 5;
    v8f acc;
    #pragma unroll
    for (int r = 0; r < 8; ++r) acc[r] = 0.0f;

    for (int t0 = 0; t0 < T; t0 += 32) {
        size_t gbase = (size_t)(b * T + t0 + tr) * Cn + h * 32 + seg * 8;
        half8 kk = *(const half8*)(k16 + gbase);
        half8 vv = *(const half8*)(v16 + gbase);
        __syncthreads();
        #pragma unroll
        for (int j = 0; j < 8; ++j) { kT[seg * 8 + j][tr] = kk[j]; vT[seg * 8 + j][tr] = vv[j]; }
        __syncthreads();
        const h16* ar = &kT[m0 + l15][0];
        half8 alo = *(const half8*)(ar + hsel * 16);
        half8 ahi = *(const half8*)(ar + hsel * 16 + 8);
        v16h af;
        #pragma unroll
        for (int j = 0; j < 8; ++j) { af[j] = alo[j]; af[j + 8] = ahi[j]; }
        v16h bf = *(const v16h*)(&vT[n0 + l15][hsel * 16]);
        acc = wmma16(af, bf, acc);
    }
    #pragma unroll
    for (int r = 0; r < 8; ++r) {
        int d = m0 + r + 8 * hsel;
        int e = n0 + l15;
        ctxT[((size_t)bh * 32 + e) * 32 + d] = (h16)acc[r];   // transposed store: [e][d]
    }
}

// ---------------------------------------------------------------- d_inv[(bh*T1+t)*4+i] = 1/(q . ksum_i)
__global__ __launch_bounds__(256) void dinv_kernel(
    const h16* __restrict__ q16, const float* __restrict__ ksum,
    float* __restrict__ dinv, int ninputs)
{
    size_t idx = (size_t)blockIdx.x * 256 + threadIdx.x; // = bh*T1 + t
    int bh = (int)(idx / T1n); int t = (int)(idx % T1n);
    int b = bh >> 3, h = bh & 7;
    const h16* qr = q16 + (size_t)(b * T1n + t) * Cn + h * 32;
    float qv[32];
    #pragma unroll
    for (int d = 0; d < 32; ++d) qv[d] = (float)qr[d];
    for (int i = 0; i < ninputs; ++i) {
        const float* ks = ksum + i * (Bn * Hn * 32) + bh * 32;
        float s = 0.0f;
        #pragma unroll
        for (int d = 0; d < 32; ++d) s += qv[d] * ks[d];
        dinv[idx * 4 + i] = 1.0f / s;
    }
}

// ---------------------------------------------------------------- apply: out = q + sum_i (q@ctx_i)*dinv_i
template <int NIN>
__global__ __launch_bounds__(256) void apply_kernel(
    const h16* __restrict__ q16, const h16* __restrict__ ctxT,
    const float* __restrict__ dinv, h16* __restrict__ out16)
{
    int bh = blockIdx.x; int b = bh >> 3, h = bh & 7;
    int wave = threadIdx.x >> 5, lane = threadIdx.x & 31;
    int hsel = lane >> 4, l15 = lane & 15;
    int t0 = (blockIdx.y * 8 + wave) * 16;
    const h16* arow = q16 + (size_t)(b * T1n + t0 + l15) * Cn + h * 32;
    half8 alo = *(const half8*)(arow + hsel * 16);
    half8 ahi = *(const half8*)(arow + hsel * 16 + 8);
    v16h af;
    #pragma unroll
    for (int j = 0; j < 8; ++j) { af[j] = alo[j]; af[j + 8] = ahi[j]; }
    v8f acc[NIN][2];
    #pragma unroll
    for (int i = 0; i < NIN; ++i)
        #pragma unroll
        for (int nt = 0; nt < 2; ++nt) {
            v8f c;
            #pragma unroll
            for (int r = 0; r < 8; ++r) c[r] = 0.0f;
            const h16* bp = ctxT + ((size_t)((i * 64 + bh) * 32 + nt * 16 + l15)) * 32 + hsel * 16;
            v16h bf = *(const v16h*)bp;
            acc[i][nt] = wmma16(af, bf, c);
        }
    #pragma unroll
    for (int nt = 0; nt < 2; ++nt)
        #pragma unroll
        for (int r = 0; r < 8; ++r) {
            int t = t0 + r + 8 * hsel;
            int e = nt * 16 + l15;
            size_t qidx = (size_t)(b * T1n + t) * Cn + h * 32 + e;
            float v = (float)q16[qidx];
            #pragma unroll
            for (int i = 0; i < NIN; ++i)
                v += acc[i][nt][r] * dinv[((size_t)bh * T1n + t) * 4 + i];
            out16[qidx] = (h16)v;
        }
}

// ---------------------------------------------------------------- gate MLP pieces
__global__ __launch_bounds__(256) void gate1_kernel(
    const float* __restrict__ pos, const float* __restrict__ w1, const float* __restrict__ b1,
    h16* __restrict__ g1)
{
    size_t idx = (size_t)blockIdx.x * 256 + threadIdx.x; // t*512 + j
    int t = (int)(idx >> 9), j = (int)(idx & 511);
    float v = pos[t * 2] * w1[j] + pos[t * 2 + 1] * w1[512 + j] + b1[j];
    g1[idx] = (h16)gelu_f(v);
}

__global__ __launch_bounds__(256) void gate3_kernel(
    const h16* __restrict__ g2, const float* __restrict__ w3, const float* __restrict__ b3,
    float* __restrict__ gate)
{
    int t = blockIdx.x * 256 + threadIdx.x;
    const h16* gr = g2 + (size_t)t * NIn;
    float o[4] = { b3[0], b3[1], b3[2], b3[3] };
    for (int k = 0; k < NIn; ++k) {
        float g = (float)gr[k];
        o[0] += g * w3[k * 4 + 0]; o[1] += g * w3[k * 4 + 1];
        o[2] += g * w3[k * 4 + 2]; o[3] += g * w3[k * 4 + 3];
    }
    float m = fmaxf(fmaxf(o[0], o[1]), fmaxf(o[2], o[3]));
    float e0 = __expf(o[0] - m), e1 = __expf(o[1] - m), e2 = __expf(o[2] - m), e3 = __expf(o[3] - m);
    float s = e0 + e1 + e2 + e3;
    gate[(size_t)t * 4 + 0] = e0 / s; gate[(size_t)t * 4 + 1] = e1 / s;
    gate[(size_t)t * 4 + 2] = e2 / s; gate[(size_t)t * 4 + 3] = e3 / s;
}

// ---------------------------------------------------------------- elementwise utils
__global__ __launch_bounds__(256) void f32_to_f16_kernel(const float* __restrict__ s, h16* __restrict__ d) {
    size_t i = (size_t)blockIdx.x * 256 + threadIdx.x; d[i] = (h16)s[i];
}
__global__ __launch_bounds__(256) void zero_f32_kernel(float* __restrict__ p) {
    size_t i = (size_t)blockIdx.x * 256 + threadIdx.x; p[i] = 0.0f;
}

// ================================================================ host orchestration
static inline void launch_gemm(int epi, const h16* A, const h16* WT, const float* bias,
                               void* out, int M, int K, int N,
                               const float* gate, int e, hipStream_t s)
{
    dim3 g(M / 128, N / 128), b(256);
    switch (epi) {
        case 0: gemm_wmma<0><<<g, b, 0, s>>>(A, WT, bias, out, M, K, N, gate, e); break;
        case 1: gemm_wmma<1><<<g, b, 0, s>>>(A, WT, bias, out, M, K, N, gate, e); break;
        case 2: gemm_wmma<2><<<g, b, 0, s>>>(A, WT, bias, out, M, K, N, gate, e); break;
        case 3: gemm_wmma<3><<<g, b, 0, s>>>(A, WT, bias, out, M, K, N, gate, e); break;
        case 4: gemm_wmma<4><<<g, b, 0, s>>>(A, WT, bias, out, M, K, N, gate, e); break;
    }
}
static inline void launch_wtrans(const float* src, h16* dst, int K, int N, hipStream_t s) {
    int total = K * N;
    wtrans_kernel<<<(total + 255) / 256, 256, 0, s>>>(src, dst, K, N);
}

extern "C" void kernel_launch(void* const* d_in, const int* in_sizes, int n_in,
                              void* d_out, int out_size, void* d_ws, size_t ws_size,
                              hipStream_t stream)
{
    const float* in_x  = (const float*)d_in[0];
    const float* in_y[3] = { (const float*)d_in[1], (const float*)d_in[2], (const float*)d_in[3] };
    const float* in_pos = (const float*)d_in[4];
    const float* ca_wq = (const float*)d_in[5];  const float* ca_bq = (const float*)d_in[6];
    const float* ca_wk = (const float*)d_in[7];  const float* ca_bk = (const float*)d_in[8];
    const float* ca_wv = (const float*)d_in[9];  const float* ca_bv = (const float*)d_in[10];
    const float* ca_wp = (const float*)d_in[11]; const float* ca_bp = (const float*)d_in[12];
    const float* sa_wq = (const float*)d_in[13]; const float* sa_bq = (const float*)d_in[14];
    const float* sa_wk = (const float*)d_in[15]; const float* sa_bk = (const float*)d_in[16];
    const float* sa_wv = (const float*)d_in[17]; const float* sa_bv = (const float*)d_in[18];
    const float* sa_wp = (const float*)d_in[19]; const float* sa_bp = (const float*)d_in[20];
    const float* m1w1 = (const float*)d_in[21];  const float* m1b1 = (const float*)d_in[22];
    const float* m1w2 = (const float*)d_in[23];  const float* m1b2 = (const float*)d_in[24];
    const float* m2w1 = (const float*)d_in[25];  const float* m2b1 = (const float*)d_in[26];
    const float* m2w2 = (const float*)d_in[27];  const float* m2b2 = (const float*)d_in[28];
    const float* gw1 = (const float*)d_in[29];   const float* gb1 = (const float*)d_in[30];
    const float* gw2 = (const float*)d_in[31];   const float* gb2 = (const float*)d_in[32];
    const float* gw3 = (const float*)d_in[33];   const float* gb3 = (const float*)d_in[34];
    const float* ln1g = (const float*)d_in[35];  const float* ln1b = (const float*)d_in[36];
    const float* ln3g = (const float*)d_in[37];  const float* ln3b = (const float*)d_in[38];
    const float* ln4g = (const float*)d_in[39];  const float* ln4b = (const float*)d_in[40];
    const float* ln5g = (const float*)d_in[41];  const float* ln5b = (const float*)d_in[42];
    const float* ln2g = (const float*)d_in[43];  const float* ln2b = (const float*)d_in[44];

    // -------- workspace carve (256B aligned)
    char* ws = (char*)d_ws; size_t off = 0;
    auto carve = [&](size_t bytes) -> char* {
        char* p = ws + off; off += (bytes + 255) & ~(size_t)255; return p;
    };
    const int CC = Cn * Cn;             // 65536
    const int CN = Cn * NIn;            // 131072
    h16* wt_ca_wq = (h16*)carve(CC * 2);
    h16* wt_ca_wk = (h16*)carve(3 * CC * 2);
    h16* wt_ca_wv = (h16*)carve(3 * CC * 2);
    h16* wt_ca_wp = (h16*)carve(CC * 2);
    h16* wt_sa_wq = (h16*)carve(CC * 2);
    h16* wt_sa_wk = (h16*)carve(CC * 2);
    h16* wt_sa_wv = (h16*)carve(CC * 2);
    h16* wt_sa_wp = (h16*)carve(CC * 2);
    h16* wt_m1w1  = (h16*)carve(4 * CN * 2);
    h16* wt_m1w2  = (h16*)carve(4 * CN * 2);
    h16* wt_m2w1  = (h16*)carve(4 * CN * 2);
    h16* wt_m2w2  = (h16*)carve(4 * CN * 2);
    h16* wt_gw2   = (h16*)carve(NIn * NIn * 2);
    h16*   bufX16 = (h16*)carve((size_t)M1n * Cn * 2);   // ln-x / attn-out f16
    h16*   bufQ   = (h16*)carve((size_t)M1n * Cn * 2);   // q16 / moe input f16
    h16*   bufK   = (h16*)carve((size_t)M1n * Cn * 2);
    h16*   bufV   = (h16*)carve((size_t)M1n * Cn * 2);
    h16*   bufY   = (h16*)carve((size_t)M2n * Cn * 2);   // ln(y_i)
    h16*   bufH   = (h16*)carve((size_t)M1n * NIn * 2);  // moe hidden / gate g1
    float* bufP   = (float*)carve((size_t)M1n * Cn * 4); // pre-softmax proj / moe acc / gate g2(f16)
    float* xbuf   = (float*)carve((size_t)M1n * Cn * 4); // residual state
    float* gateb  = (float*)carve((size_t)M1n * 4 * 4);
    h16*   ctxTb  = (h16*)carve(3 * 64 * 1024 * 2);
    float* ksumb  = (float*)carve(3 * 64 * 32 * 4);
    float* dinvb  = (float*)carve((size_t)64 * T1n * 4 * 4);
    (void)in_sizes; (void)n_in; (void)out_size; (void)ws_size;

    // -------- 1. weight transposes (f32 [K,N] -> f16 [N,K])
    launch_wtrans(ca_wq, wt_ca_wq, Cn, Cn, stream);
    for (int i = 0; i < 3; ++i) launch_wtrans(ca_wk + (size_t)i * CC, wt_ca_wk + (size_t)i * CC, Cn, Cn, stream);
    for (int i = 0; i < 3; ++i) launch_wtrans(ca_wv + (size_t)i * CC, wt_ca_wv + (size_t)i * CC, Cn, Cn, stream);
    launch_wtrans(ca_wp, wt_ca_wp, Cn, Cn, stream);
    launch_wtrans(sa_wq, wt_sa_wq, Cn, Cn, stream);
    launch_wtrans(sa_wk, wt_sa_wk, Cn, Cn, stream);
    launch_wtrans(sa_wv, wt_sa_wv, Cn, Cn, stream);
    launch_wtrans(sa_wp, wt_sa_wp, Cn, Cn, stream);
    for (int e = 0; e < 4; ++e) {
        launch_wtrans(m1w1 + (size_t)e * CN, wt_m1w1 + (size_t)e * CN, Cn, NIn, stream);
        launch_wtrans(m1w2 + (size_t)e * CN, wt_m1w2 + (size_t)e * CN, NIn, Cn, stream);
        launch_wtrans(m2w1 + (size_t)e * CN, wt_m2w1 + (size_t)e * CN, Cn, NIn, stream);
        launch_wtrans(m2w2 + (size_t)e * CN, wt_m2w2 + (size_t)e * CN, NIn, Cn, stream);
    }
    launch_wtrans(gw2, wt_gw2, NIn, NIn, stream);

    // -------- 2. residual state init
    hipMemcpyAsync(xbuf, in_x, (size_t)M1n * Cn * 4, hipMemcpyDeviceToDevice, stream);

    // -------- 3. gate network
    gate1_kernel<<<(M1n * NIn) / 256, 256, 0, stream>>>(in_pos, gw1, gb1, bufH);
    launch_gemm(2, bufH, wt_gw2, gb2, (h16*)bufP, M1n, NIn, NIn, nullptr, 0, stream);
    gate3_kernel<<<M1n / 256, 256, 0, stream>>>((h16*)bufP, gw3, gb3, gateb);

    // -------- 4. cross attention
    ln_f16_kernel<<<M1n / 8, 256, 0, stream>>>(in_x, ln1g, ln1b, bufX16);
    launch_gemm(0, bufX16, wt_ca_wq, ca_bq, bufP, M1n, Cn, Cn, nullptr, 0, stream);
    softmax32_kernel<<<M1n, 256, 0, stream>>>(bufP, bufQ);
    for (int i = 0; i < 3; ++i) {
        ln_f16_kernel<<<M2n / 8, 256, 0, stream>>>(in_y[i], ln2g + 256 * i, ln2b + 256 * i, bufY);
        launch_gemm(0, bufY, wt_ca_wk + (size_t)i * CC, ca_bk + 256 * i, bufP, M2n, Cn, Cn, nullptr, 0, stream);
        softmax32_kernel<<<M2n, 256, 0, stream>>>(bufP, bufK);
        launch_gemm(1, bufY, wt_ca_wv + (size_t)i * CC, ca_bv + 256 * i, bufV, M2n, Cn, Cn, nullptr, 0, stream);
        ksum_kernel<<<64, 256, 0, stream>>>(bufK, ksumb + i * 2048, T2n);
        ctx_kernel<<<64, 128, 0, stream>>>(bufK, bufV, ctxTb + (size_t)i * 65536, T2n);
    }
    dinv_kernel<<<(64 * T1n) / 256, 256, 0, stream>>>(bufQ, ksumb, dinvb, 3);
    apply_kernel<3><<<dim3(64, T1n / 128), 256, 0, stream>>>(bufQ, ctxTb, dinvb, bufX16);
    launch_gemm(3, bufX16, wt_ca_wp, ca_bp, xbuf, M1n, Cn, Cn, nullptr, 0, stream);

    // -------- 5. MoE 1
    f32_to_f16_kernel<<<(M1n * Cn) / 256, 256, 0, stream>>>(xbuf, bufQ);
    zero_f32_kernel<<<(M1n * Cn) / 256, 256, 0, stream>>>(bufP);
    for (int e = 0; e < 4; ++e) {
        launch_gemm(2, bufQ, wt_m1w1 + (size_t)e * CN, m1b1 + 512 * e, bufH, M1n, Cn, NIn, nullptr, 0, stream);
        launch_gemm(4, bufH, wt_m1w2 + (size_t)e * CN, m1b2 + 256 * e, bufP, M1n, NIn, Cn, gateb, e, stream);
    }
    ln_add_kernel<<<M1n / 8, 256, 0, stream>>>(bufP, xbuf, ln3g, ln3b, xbuf);

    // -------- 6. self attention
    ln_f16_kernel<<<M1n / 8, 256, 0, stream>>>(xbuf, ln4g, ln4b, bufX16);
    launch_gemm(0, bufX16, wt_sa_wq, sa_bq, bufP, M1n, Cn, Cn, nullptr, 0, stream);
    softmax32_kernel<<<M1n, 256, 0, stream>>>(bufP, bufQ);
    launch_gemm(0, bufX16, wt_sa_wk, sa_bk, bufP, M1n, Cn, Cn, nullptr, 0, stream);
    softmax32_kernel<<<M1n, 256, 0, stream>>>(bufP, bufK);
    launch_gemm(1, bufX16, wt_sa_wv, sa_bv, bufV, M1n, Cn, Cn, nullptr, 0, stream);
    ksum_kernel<<<64, 256, 0, stream>>>(bufK, ksumb, T1n);
    ctx_kernel<<<64, 128, 0, stream>>>(bufK, bufV, ctxTb, T1n);
    dinv_kernel<<<(64 * T1n) / 256, 256, 0, stream>>>(bufQ, ksumb, dinvb, 1);
    apply_kernel<1><<<dim3(64, T1n / 128), 256, 0, stream>>>(bufQ, ctxTb, dinvb, bufX16);
    launch_gemm(3, bufX16, wt_sa_wp, sa_bp, xbuf, M1n, Cn, Cn, nullptr, 0, stream);

    // -------- 7. MoE 2 (final LN-add writes d_out)
    f32_to_f16_kernel<<<(M1n * Cn) / 256, 256, 0, stream>>>(xbuf, bufQ);
    zero_f32_kernel<<<(M1n * Cn) / 256, 256, 0, stream>>>(bufP);
    for (int e = 0; e < 4; ++e) {
        launch_gemm(2, bufQ, wt_m2w1 + (size_t)e * CN, m2b1 + 512 * e, bufH, M1n, Cn, NIn, nullptr, 0, stream);
        launch_gemm(4, bufH, wt_m2w2 + (size_t)e * CN, m2b2 + 256 * e, bufP, M1n, NIn, Cn, gateb, e, stream);
    }
    ln_add_kernel<<<M1n / 8, 256, 0, stream>>>(bufP, xbuf, ln5g, ln5b, (float*)d_out);
}
